// VectorQuantizer_15006615733662
// MI455X (gfx1250) — compile-verified, matrix operand order checked by reading, not run on hardware
//
#include <hip/hip_runtime.h>
#include <stdint.h>

typedef __attribute__((ext_vector_type(16))) __bf16 v16bf;
typedef __attribute__((ext_vector_type(8)))  float  v8f;

#define NUM_EMB   512
#define EDIM      64
#define LDS_STRIDE 72                 // bf16 elems per codebook row (144B, conflict-free)
#define NROWS     (16*128*128)        // 262144
#define NTILES    (NROWS/16)          // 16384
#define LOSS_OFF  ((size_t)NROWS*EDIM) // 16777216
#define IDX_OFF   (LOSS_OFF + 1)
#define NBLOCKS   1024

// float -> bf16 with round-to-nearest-even (no NaN in this data)
__device__ __forceinline__ unsigned f2bf(float f) {
    union { float f; unsigned u; } c; c.f = f;
    unsigned u = c.u;
    return (u + 0x7fffu + ((u >> 16) & 1u)) >> 16;
}
__device__ __forceinline__ unsigned pk2(float a, float b) {
    return f2bf(a) | (f2bf(b) << 16);
}

union V16 { v16bf v; uint4 q[2]; unsigned u[8]; };

__global__ __launch_bounds__(256) void vq_main(const float* __restrict__ z,
                                               const float* __restrict__ emb,
                                               float* __restrict__ out,
                                               float* __restrict__ ws) {
    __shared__ unsigned short eb[NUM_EMB * LDS_STRIDE]; // (-2*emb) in bf16
    __shared__ float nrm[NUM_EMB];                      // ||e||^2 exact f32
    __shared__ float wsum[8];

    const int tid = threadIdx.x;

    // ---- one-time codebook stage: 2 rows per thread (512 rows / 256 thr) ----
    {
        #pragma unroll
        for (int rr = 0; rr < 2; ++rr) {
            const int r = tid * 2 + rr;
            const float* er = emb + (size_t)r * EDIM;
            unsigned* dst = (unsigned*)&eb[r * LDS_STRIDE];
            float s = 0.f;
            #pragma unroll
            for (int i = 0; i < EDIM; i += 2) {
                float a = er[i], b = er[i + 1];
                s += a * a + b * b;
                dst[i / 2] = pk2(-2.f * a, -2.f * b);
            }
            nrm[r] = s;
        }
    }
    __syncthreads();

    const int lane      = tid & 31;
    const int waveInBlk = tid >> 5;
    const int q    = lane & 15;   // row (A) / code column (B) within tile
    const int half = lane >> 4;
    const int kb   = half * 8;    // K sub-run base per 16-bit WMMA layout
    const int waveGlobal = blockIdx.x * 8 + waveInBlk;
    const int totWaves   = NBLOCKS * 8;

    float sumsq = 0.f;

    for (int t = waveGlobal; t < NTILES; t += totWaves) {
        const int row0 = t * 16;

        // ---- load A tile (16 rows x 64), pack to bf16 in WMMA layout ----
        const float4* zr4 = (const float4*)(z + (size_t)(row0 + q) * EDIM);
        float4 f0 = zr4[(kb)      / 4], f1 = zr4[(kb)      / 4 + 1];
        float4 f2 = zr4[(16 + kb) / 4], f3 = zr4[(16 + kb) / 4 + 1];
        float4 f4 = zr4[(32 + kb) / 4], f5 = zr4[(32 + kb) / 4 + 1];
        float4 f6 = zr4[(48 + kb) / 4], f7 = zr4[(48 + kb) / 4 + 1];
        V16 A0, A1;
        A0.u[0] = pk2(f0.x, f0.y); A0.u[1] = pk2(f0.z, f0.w);
        A0.u[2] = pk2(f1.x, f1.y); A0.u[3] = pk2(f1.z, f1.w);
        A0.u[4] = pk2(f2.x, f2.y); A0.u[5] = pk2(f2.z, f2.w);
        A0.u[6] = pk2(f3.x, f3.y); A0.u[7] = pk2(f3.z, f3.w);
        A1.u[0] = pk2(f4.x, f4.y); A1.u[1] = pk2(f4.z, f4.w);
        A1.u[2] = pk2(f5.x, f5.y); A1.u[3] = pk2(f5.z, f5.w);
        A1.u[4] = pk2(f6.x, f6.y); A1.u[5] = pk2(f6.z, f6.w);
        A1.u[6] = pk2(f7.x, f7.y); A1.u[7] = pk2(f7.z, f7.w);

        float minv[8]; int mini[8];
        #pragma unroll
        for (int s = 0; s < 8; ++s) { minv[s] = 3.4e38f; mini[s] = 0; }

        // ---- scan all 512 codes, 32 at a time with DUAL accumulators so the
        //      two WMMA dep-chains interleave and argmin VALU fills the
        //      WMMA->VALU hazard slots ----
        #pragma unroll 2
        for (int c0 = 0; c0 < NUM_EMB; c0 += 32) {
            const int ca = c0 + q;
            const int cb = c0 + 16 + q;
            const float nva = nrm[ca];
            const float nvb = nrm[cb];
            v8f acc0, acc1;
            #pragma unroll
            for (int s = 0; s < 8; ++s) { acc0[s] = nva; acc1[s] = nvb; }

            const uint4* la = (const uint4*)&eb[ca * LDS_STRIDE];
            const uint4* lb = (const uint4*)&eb[cb * LDS_STRIDE];
            V16 B0a, B1a, B0b, B1b;
            B0a.q[0] = la[(kb)      / 8];
            B0a.q[1] = la[(16 + kb) / 8];
            B1a.q[0] = la[(32 + kb) / 8];
            B1a.q[1] = la[(48 + kb) / 8];
            B0b.q[0] = lb[(kb)      / 8];
            B0b.q[1] = lb[(16 + kb) / 8];
            B1b.q[0] = lb[(32 + kb) / 8];
            B1b.q[1] = lb[(48 + kb) / 8];

            acc0 = __builtin_amdgcn_wmma_f32_16x16x32_bf16(false, A0.v, false, B0a.v,
                                                           (short)0, acc0, false, false);
            acc1 = __builtin_amdgcn_wmma_f32_16x16x32_bf16(false, A0.v, false, B0b.v,
                                                           (short)0, acc1, false, false);
            acc0 = __builtin_amdgcn_wmma_f32_16x16x32_bf16(false, A1.v, false, B1a.v,
                                                           (short)0, acc0, false, false);
            acc1 = __builtin_amdgcn_wmma_f32_16x16x32_bf16(false, A1.v, false, B1b.v,
                                                           (short)0, acc1, false, false);

            #pragma unroll
            for (int s = 0; s < 8; ++s) {
                float v0 = acc0[s];
                if (v0 < minv[s]) { minv[s] = v0; mini[s] = ca; }
            }
            #pragma unroll
            for (int s = 0; s < 8; ++s) {
                float v1 = acc1[s];
                if (v1 < minv[s]) { minv[s] = v1; mini[s] = cb; }
            }
        }

        // ---- cross-lane argmin over the 16 lanes sharing each row-half ----
        #pragma unroll
        for (int s = 0; s < 8; ++s) {
            float v = minv[s]; int i = mini[s];
            #pragma unroll
            for (int m = 1; m <= 8; m <<= 1) {
                float ov = __shfl_xor(v, m, 32);
                int   oi = __shfl_xor(i, m, 32);
                if (ov < v || (ov == v && oi < i)) { v = ov; i = oi; }
            }
            minv[s] = v; mini[s] = i;
        }

        // lanes 0 and 16 write indices for rows 0-7 / 8-15 (as float values)
        if (q == 0) {
            #pragma unroll
            for (int s = 0; s < 8; ++s)
                out[IDX_OFF + row0 + half * 8 + s] = (float)mini[s];
        }

        // ---- gather z_q from exact f32 codebook, write out, local loss ----
        #pragma unroll
        for (int s = 0; s < 8; ++s) {
            const int row = row0 + half * 8 + s;
            const float4 e4 = ((const float4*)(emb + (size_t)mini[s] * EDIM))[q];
            const float4 z4 = ((const float4*)(z   + (size_t)row     * EDIM))[q];
            ((float4*)(out + (size_t)row * EDIM))[q] = e4;
            float dx = e4.x - z4.x, dy = e4.y - z4.y;
            float dz = e4.z - z4.z, dw = e4.w - z4.w;
            sumsq += dx * dx + dy * dy + dz * dz + dw * dw;
        }
    }

    // deterministic block-level loss partial
    #pragma unroll
    for (int m = 1; m < 32; m <<= 1) sumsq += __shfl_xor(sumsq, m, 32);
    if (lane == 0) wsum[waveInBlk] = sumsq;
    __syncthreads();
    if (tid == 0) {
        float s = 0.f;
        #pragma unroll
        for (int w = 0; w < 8; ++w) s += wsum[w];
        ws[blockIdx.x] = s;
    }
}

__global__ __launch_bounds__(256) void vq_loss(const float* __restrict__ ws,
                                               float* __restrict__ out) {
    __shared__ float red[256];
    float s = 0.f;
    for (int i = threadIdx.x; i < NBLOCKS; i += 256) s += ws[i];
    red[threadIdx.x] = s;
    __syncthreads();
    for (int w = 128; w > 0; w >>= 1) {
        if (threadIdx.x < w) red[threadIdx.x] += red[threadIdx.x + w];
        __syncthreads();
    }
    if (threadIdx.x == 0)
        out[LOSS_OFF] = red[0] * (1.25f / 16777216.f);
}

extern "C" void kernel_launch(void* const* d_in, const int* in_sizes, int n_in,
                              void* d_out, int out_size, void* d_ws, size_t ws_size,
                              hipStream_t stream) {
    const float* z   = (const float*)d_in[0];
    const float* emb = (const float*)d_in[1];
    float* out = (float*)d_out;
    float* ws  = (float*)d_ws;
    vq_main<<<NBLOCKS, 256, 0, stream>>>(z, emb, out, ws);
    vq_loss<<<1, 256, 0, stream>>>(ws, out);
}
